// SelfAttention_55207509623112
// MI455X (gfx1250) — matrix operand order repaired and checked by reading, not compiled, not run on hardware
//
#include <hip/hip_runtime.h>

typedef __bf16 bf16;
typedef __attribute__((ext_vector_type(16))) __bf16 v16bf;
typedef __attribute__((ext_vector_type(8)))  float  v8f;

#define D_DIM 1024
#define T_DIM 2048
#define B_DIM 4
#define H_DIM 16
#define S_DIM 64
#define M_ROWS (B_DIM * T_DIM) /* 8192 */

// ---------------------------------------------------------------------------
// CDNA5 async global->LDS copy (ASYNCcnt path, no VGPR round trip).
// LDS address from the VDST VGPR plus offset; GV mode: 64-bit vaddr, saddr = off.
// ---------------------------------------------------------------------------
__device__ inline void async_b128(uint32_t lds_off, const void* gptr) {
  unsigned long long ga = (unsigned long long)(uintptr_t)gptr;
  asm volatile("global_load_async_to_lds_b128 %0, %1, off"
               :: "v"(lds_off), "v"(ga) : "memory");
}
__device__ inline void wait_async0() {
  asm volatile("s_wait_asynccnt 0x0" ::: "memory");
}
__device__ inline uint32_t lds_off32(const void* p) {
  return (uint32_t)(uintptr_t)p;  // low 32 bits of generic addr = LDS offset
}

// ---------------------------------------------------------------------------
// WMMA helpers — fragment loads follow the CDNA5 ISA VGPR layouts (CDNA5 ISA)
// ---------------------------------------------------------------------------

// A-matrix 16x32 bf16 (MxK). Lanes 0-15: M=lane, K=0-7 & 16-23.
// Lanes 16-31: M=lane-16, K=8-15 & 24-31. Two 16B chunks per lane.
__device__ inline v16bf frag_a_load(const bf16* __restrict__ base, int ldm) {
  const int lane = threadIdx.x & 31;
  const int m  = lane & 15;
  const int k0 = (lane & 16) ? 8 : 0;
  union { v16bf v; uint4 q[2]; } u;
  u.q[0] = *(const uint4*)(base + m * ldm + k0);
  u.q[1] = *(const uint4*)(base + m * ldm + k0 + 16);
  return u.v;
}

// B-matrix 32x16 bf16 (KxN), stored in memory as [n][k] (row = N, contiguous K).
// Lanes 0-15: N=lane, K=0-7 & 16-23. Lanes 16-31: N=lane-16, K=8-15 & 24-31.
__device__ inline v16bf frag_b_load(const bf16* __restrict__ base, int ldn) {
  const int lane = threadIdx.x & 31;
  const int n  = lane & 15;
  const int k0 = (lane & 16) ? 8 : 0;
  union { v16bf v; uint4 q[2]; } u;
  u.q[0] = *(const uint4*)(base + n * ldn + k0);
  u.q[1] = *(const uint4*)(base + n * ldn + k0 + 16);
  return u.v;
}

__device__ inline v8f wmma_bf16(v16bf a, v16bf b, v8f c) {
  // (neg_a, A, neg_b, B, c_mod, C, reuse_a, reuse_b)
  return __builtin_amdgcn_wmma_f32_16x16x32_bf16(false, a, false, b,
                                                 (short)0, c, false, false);
}

// ---------------------------------------------------------------------------
// fp32 -> bf16 conversion (grid-stride)
// ---------------------------------------------------------------------------
__global__ __launch_bounds__(256) void cvt_f32_bf16(const float* __restrict__ src,
                                                    bf16* __restrict__ dst, int n) {
  int i = blockIdx.x * blockDim.x + threadIdx.x;
  int stride = gridDim.x * blockDim.x;
  for (; i < n; i += stride) dst[i] = (bf16)src[i];
}

// ---------------------------------------------------------------------------
// Generic WMMA GEMM: out[m,n] = sum_k A[m,k] * W[n,k]   (torch Linear, x @ W.T)
// A: [M_ROWS,1024] bf16 row-major, W: [1024,1024] bf16 row-major.
// mode 0: bf16 out row-major; mode 1: bf16 out transposed Vt[b,h,s,t];
// mode 2: fp32 out + bias.
// Block tile 128x128, 8 waves (4x2) of 32x64.
// K staged via double-buffered async global->LDS copies.
// ---------------------------------------------------------------------------
__global__ __launch_bounds__(256) void gemm_xwT(const bf16* __restrict__ A,
                                                const bf16* __restrict__ W,
                                                bf16* __restrict__ out_bf,
                                                float* __restrict__ out_f,
                                                const float* __restrict__ bias,
                                                int mode) {
  __shared__ __align__(16) bf16 sA[2][128 * 40];
  __shared__ __align__(16) bf16 sB[2][128 * 40];

  const int tid  = threadIdx.x;
  const int lane = tid & 31;
  const int wave = tid >> 5;
  const int wm   = wave >> 1;      // 0..3
  const int wn   = wave & 1;       // 0..1
  const int m0   = blockIdx.x * 128;
  const int n0   = blockIdx.y * 128;
  const int halfsel = (lane >> 4) & 1;

  const bf16* gA = A + (size_t)m0 * D_DIM;
  const bf16* gW = W + (size_t)n0 * D_DIM;

  // This thread's two 16B chunks of the 128x32 tile (512 chunks / 256 threads)
  const int c0row = (tid) >> 2,        c0k = (tid & 3) * 8;
  const int c1row = (tid + 256) >> 2,  c1k = ((tid + 256) & 3) * 8;

  v8f zero = {};
  v8f acc[2][4];
  for (int i = 0; i < 2; ++i)
    for (int j = 0; j < 4; ++j) acc[i][j] = zero;

  // Prologue: async-stage tile kk=0 into buffer 0
  async_b128(lds_off32(&sA[0][c0row * 40 + c0k]), gA + (size_t)c0row * D_DIM + c0k);
  async_b128(lds_off32(&sB[0][c0row * 40 + c0k]), gW + (size_t)c0row * D_DIM + c0k);
  async_b128(lds_off32(&sA[0][c1row * 40 + c1k]), gA + (size_t)c1row * D_DIM + c1k);
  async_b128(lds_off32(&sB[0][c1row * 40 + c1k]), gW + (size_t)c1row * D_DIM + c1k);

  int buf = 0;
  for (int kk = 0; kk < D_DIM; kk += 32) {
    wait_async0();          // my wave's copies for this tile done
    __syncthreads();        // everyone's copies done; prior buffer free

    if (kk + 32 < D_DIM) {  // prefetch next K-slab into the other buffer
      const int nb = buf ^ 1;
      const int kn = kk + 32;
      async_b128(lds_off32(&sA[nb][c0row * 40 + c0k]), gA + (size_t)c0row * D_DIM + kn + c0k);
      async_b128(lds_off32(&sB[nb][c0row * 40 + c0k]), gW + (size_t)c0row * D_DIM + kn + c0k);
      async_b128(lds_off32(&sA[nb][c1row * 40 + c1k]), gA + (size_t)c1row * D_DIM + kn + c1k);
      async_b128(lds_off32(&sB[nb][c1row * 40 + c1k]), gW + (size_t)c1row * D_DIM + kn + c1k);
      __builtin_prefetch(gA + (size_t)(wm * 32) * D_DIM + kn + 64, 0, 1);
    }

    v16bf af[2], bfv[4];
    for (int mt = 0; mt < 2; ++mt)
      af[mt] = frag_a_load(&sA[buf][(wm * 32 + mt * 16) * 40], 40);
    for (int nt = 0; nt < 4; ++nt)
      bfv[nt] = frag_b_load(&sB[buf][(wn * 64 + nt * 16) * 40], 40);
    for (int mt = 0; mt < 2; ++mt)
      for (int nt = 0; nt < 4; ++nt)
        acc[mt][nt] = wmma_bf16(af[mt], bfv[nt], acc[mt][nt]);

    buf ^= 1;
  }

  // Epilogue: C/D layout -> memory
  for (int mt = 0; mt < 2; ++mt) {
    for (int nt = 0; nt < 4; ++nt) {
      for (int r = 0; r < 8; ++r) {
        const int mrow = m0 + wm * 32 + mt * 16 + r + 8 * halfsel;
        const int col  = n0 + wn * 64 + nt * 16 + (lane & 15);
        const float v  = acc[mt][nt][r];
        if (mode == 0) {
          out_bf[(size_t)mrow * D_DIM + col] = (bf16)v;
        } else if (mode == 1) {
          const int b = mrow >> 11, t = mrow & (T_DIM - 1);
          const int h = col >> 6,  s = col & (S_DIM - 1);
          out_bf[((size_t)(b * H_DIM + h) * S_DIM + s) * T_DIM + t] = (bf16)v;
        } else {
          out_f[(size_t)mrow * D_DIM + col] = v + bias[col];
        }
      }
    }
  }
}

// ---------------------------------------------------------------------------
// Flash attention (causal, online softmax).
// Grid: (T/64, B*H). Block: 128 threads = 4 waves; wave handles 16 query rows.
// Q,K: [B*T,1024] bf16 (head slice at col h*64). Vt: [B,H,64,T] bf16.
// K/V^T tiles double-buffered via async global->LDS copies.
// ---------------------------------------------------------------------------
__global__ __launch_bounds__(128) void attn_kernel(const bf16* __restrict__ Q,
                                                   const bf16* __restrict__ K,
                                                   const bf16* __restrict__ Vt,
                                                   bf16* __restrict__ Y) {
  __shared__ __align__(16) bf16 lK[2][64 * 72];    // [j][d] key tile
  __shared__ __align__(16) bf16 lV[2][64 * 72];    // [d][j] value^T tile
  __shared__ __align__(16) bf16 lP[4][16 * 72];    // per-wave probs [m][j]

  const int tid  = threadIdx.x;
  const int lane = tid & 31;
  const int wave = tid >> 5;
  const int halfsel = (lane >> 4) & 1;
  const int qtile = blockIdx.x;
  const int bh    = blockIdx.y;
  const int b = bh >> 4, h = bh & 15;

  const int q0 = qtile * 64 + wave * 16;                 // first query row (in T)
  const size_t head_base = (size_t)b * T_DIM * D_DIM + (size_t)h * S_DIM;
  const bf16* VtB = Vt + (size_t)bh * S_DIM * T_DIM;

  // Q fragments held in registers for the whole kernel (d split 0..31 / 32..63)
  v16bf qf[2];
  qf[0] = frag_a_load(Q + head_base + (size_t)q0 * D_DIM + 0,  D_DIM);
  qf[1] = frag_a_load(Q + head_base + (size_t)q0 * D_DIM + 32, D_DIM);

  // Async-stage one (K, V^T) key tile: 64 rows x 8 chunks = 512 / 128 thr = 4 ea.
  auto issue_tiles = [&](int jt, int bsel) {
    const int j0 = jt * 64;
#pragma unroll
    for (int i = 0; i < 4; ++i) {
      const int c = tid + i * 128;
      const int row = c >> 3, kc = (c & 7) * 8;
      async_b128(lds_off32(&lK[bsel][row * 72 + kc]),
                 K + head_base + (size_t)(j0 + row) * D_DIM + kc);
      async_b128(lds_off32(&lV[bsel][row * 72 + kc]),
                 VtB + (size_t)row * T_DIM + j0 + kc);
    }
  };

  v8f zero = {};
  v8f o[4];
  for (int i = 0; i < 4; ++i) o[i] = zero;
  float m_i[8], l_i[8];
  for (int r = 0; r < 8; ++r) { m_i[r] = -1e30f; l_i[r] = 0.0f; }

  const float scale = 0.125f;  // 1/sqrt(64)

  issue_tiles(0, 0);
  int buf = 0;

  for (int jt = 0; jt <= qtile; ++jt) {           // causal: only jt <= qtile
    const int j0 = jt * 64;
    wait_async0();
    __syncthreads();
    if (jt < qtile) issue_tiles(jt + 1, buf ^ 1); // overlap copy with compute

    // S = Q K^T / sqrt(d): 4 column subtiles x 2 WMMAs over d
    v8f sc[4];
    for (int nt = 0; nt < 4; ++nt) {
      v8f s = zero;
      s = wmma_bf16(qf[0], frag_b_load(&lK[buf][nt * 16 * 72 + 0],  72), s);
      s = wmma_bf16(qf[1], frag_b_load(&lK[buf][nt * 16 * 72 + 32], 72), s);
      sc[nt] = s;
    }

    // Scale + causal mask + per-row max (C layout: row = r + 8*halfsel)
    float tmax[8];
    for (int r = 0; r < 8; ++r) tmax[r] = -1e30f;
    for (int nt = 0; nt < 4; ++nt) {
      const int jcol = j0 + nt * 16 + (lane & 15);
      for (int r = 0; r < 8; ++r) {
        const int qrow = q0 + r + 8 * halfsel;
        float v = sc[nt][r] * scale;
        v = (jcol <= qrow) ? v : -1e30f;
        sc[nt][r] = v;
        tmax[r] = fmaxf(tmax[r], v);
      }
    }
    for (int off = 1; off < 16; off <<= 1)
      for (int r = 0; r < 8; ++r)
        tmax[r] = fmaxf(tmax[r], __shfl_xor(tmax[r], off, 32));

    // Online softmax update
    float corr[8], rsum[8];
    for (int r = 0; r < 8; ++r) {
      const float mnew = fmaxf(m_i[r], tmax[r]);
      corr[r] = __expf(m_i[r] - mnew);
      m_i[r] = mnew;
      rsum[r] = 0.0f;
    }
    for (int nt = 0; nt < 4; ++nt) {
      for (int r = 0; r < 8; ++r) {
        const float p = __expf(sc[nt][r] - m_i[r]);
        rsum[r] += p;
        // C-layout -> [m][j] in per-wave LDS (re-layout for the A operand)
        lP[wave][(r + 8 * halfsel) * 72 + nt * 16 + (lane & 15)] = (bf16)p;
      }
    }
    for (int off = 1; off < 16; off <<= 1)
      for (int r = 0; r < 8; ++r) rsum[r] += __shfl_xor(rsum[r], off, 32);
    for (int r = 0; r < 8; ++r) l_i[r] = l_i[r] * corr[r] + rsum[r];
    for (int dt = 0; dt < 4; ++dt)
      for (int r = 0; r < 8; ++r) o[dt][r] = o[dt][r] * corr[r];

    // O += P @ V : A = P (16 x 32 over j), B = V^T [d][j]
    for (int dt = 0; dt < 4; ++dt) {
      for (int jk = 0; jk < 64; jk += 32) {
        v16bf pa = frag_a_load(&lP[wave][0] + jk, 72);
        v16bf vb = frag_b_load(&lV[buf][dt * 16 * 72 + jk], 72);
        o[dt] = wmma_bf16(pa, vb, o[dt]);
      }
    }
    buf ^= 1;
  }

  // Normalize and write Y[b, t, h*64 + d] (bf16)
  for (int r = 0; r < 8; ++r) {
    const float inv = 1.0f / l_i[r];
    const int trow = q0 + r + 8 * halfsel;
    const size_t obase = ((size_t)b * T_DIM + trow) * D_DIM + (size_t)h * S_DIM;
    for (int dt = 0; dt < 4; ++dt)
      Y[obase + dt * 16 + (lane & 15)] = (bf16)(o[dt][r] * inv);
  }
}

// ---------------------------------------------------------------------------
// Host-side orchestration (graph-capture safe: stream-only, ws scratch)
// ---------------------------------------------------------------------------
extern "C" void kernel_launch(void* const* d_in, const int* in_sizes, int n_in,
                              void* d_out, int out_size, void* d_ws, size_t ws_size,
                              hipStream_t stream) {
  (void)in_sizes; (void)n_in; (void)out_size; (void)ws_size;
  const float* x  = (const float*)d_in[0];
  const float* Wk = (const float*)d_in[1];
  const float* Wq = (const float*)d_in[2];
  const float* Wv = (const float*)d_in[3];
  const float* Wu = (const float*)d_in[4];
  const float* bu = (const float*)d_in[5];
  float* out = (float*)d_out;

  const size_t SZ_X = (size_t)M_ROWS * D_DIM;   // 8.39M elements
  const size_t SZ_W = (size_t)D_DIM * D_DIM;    // 1.05M elements

  char* ws = (char*)d_ws;
  bf16* xbf  = (bf16*)ws; ws += SZ_X * sizeof(bf16);
  bf16* wkbf = (bf16*)ws; ws += SZ_W * sizeof(bf16);
  bf16* wqbf = (bf16*)ws; ws += SZ_W * sizeof(bf16);
  bf16* wvbf = (bf16*)ws; ws += SZ_W * sizeof(bf16);
  bf16* wubf = (bf16*)ws; ws += SZ_W * sizeof(bf16);
  bf16* qbf  = (bf16*)ws; ws += SZ_X * sizeof(bf16);
  bf16* kbf  = (bf16*)ws; ws += SZ_X * sizeof(bf16);
  bf16* vt   = (bf16*)ws; ws += SZ_X * sizeof(bf16);
  bf16* ybf  = (bf16*)ws; ws += SZ_X * sizeof(bf16);

  cvt_f32_bf16<<<2048, 256, 0, stream>>>(x,  xbf,  (int)SZ_X);
  cvt_f32_bf16<<<512,  256, 0, stream>>>(Wk, wkbf, (int)SZ_W);
  cvt_f32_bf16<<<512,  256, 0, stream>>>(Wq, wqbf, (int)SZ_W);
  cvt_f32_bf16<<<512,  256, 0, stream>>>(Wv, wvbf, (int)SZ_W);
  cvt_f32_bf16<<<512,  256, 0, stream>>>(Wu, wubf, (int)SZ_W);

  dim3 ggrid(M_ROWS / 128, D_DIM / 128);            // 64 x 8
  gemm_xwT<<<ggrid, 256, 0, stream>>>(xbf, wqbf, qbf, nullptr, nullptr, 0);
  gemm_xwT<<<ggrid, 256, 0, stream>>>(xbf, wkbf, kbf, nullptr, nullptr, 0);
  gemm_xwT<<<ggrid, 256, 0, stream>>>(xbf, wvbf, vt,  nullptr, nullptr, 1);

  dim3 agrid(T_DIM / 64, B_DIM * H_DIM);            // 32 x 64
  attn_kernel<<<agrid, 128, 0, stream>>>(qbf, kbf, vt, ybf);

  gemm_xwT<<<ggrid, 256, 0, stream>>>(ybf, wubf, nullptr, out, bu, 2);
}
